// MolecularEnergyPredictor_64939905515523
// MI455X (gfx1250) — matrix operand order repaired and checked by reading, not compiled, not run on hardware
//
#include <hip/hip_runtime.h>
#include <math.h>

#define N_ATOMS 1024
#define NRBF    80

typedef __attribute__((ext_vector_type(16))) __bf16 v16bf;
typedef __attribute__((ext_vector_type(8)))  float  v8f;

__device__ __forceinline__ float sspf(float x){
  // shifted softplus: log(1+e^x) - log(2), numerically stable
  float ax = fabsf(x);
  float r  = __logf(1.0f + __expf(-ax));
  return fmaxf(x, 0.0f) + r - 0.69314718056f;
}

__device__ __forceinline__ __bf16 f2bf(float f){ return (__bf16)f; }

// -------------------------------------------------------------------------
// Embedding: s0[i,u] = emb[u, z[i]]   (one-hot matmul == column gather)
// -------------------------------------------------------------------------
__global__ void embed_kernel(const float* __restrict__ emb,
                             const int*   __restrict__ z,
                             float* __restrict__ s0){
  int t = blockIdx.x * blockDim.x + threadIdx.x;
  if (t >= N_ATOMS * 32) return;
  int i = t >> 5, u = t & 31;
  s0[t] = emb[u * 5 + z[i]];
}

// -------------------------------------------------------------------------
// Fused convolution layer.
//   Per wave-task (atom i, j-tile of 16):
//     geometry -> 80 RBF (bf16, B-layout) -> WMMA dense(80->C) -> ssp
//     -> shfl repack -> WMMA dense(C->C)  (x2 radials R0,R1)
//     -> per-pair message accumulation (0x0, 0x1, 1x0, 1x1 dot, 1x1 cross)
//   8 waves / block cover all 64 j-tiles for atom i = blockIdx.x.
//   Butterfly-reduce over the 16 pair-lanes, atomicAdd into raw outputs.
// -------------------------------------------------------------------------
template<int CIN, bool FIRST>
__global__ __launch_bounds__(256, 1) void conv_kernel(
    const float* __restrict__ r,
    const float* __restrict__ s_in,
    const float* __restrict__ v_in,
    const float* __restrict__ w1a, const float* __restrict__ b1a,
    const float* __restrict__ w2a, const float* __restrict__ b2a,
    const float* __restrict__ w1b, const float* __restrict__ b1b,
    const float* __restrict__ w2b, const float* __restrict__ b2b,
    float* __restrict__ out_s, float* __restrict__ out_v)
{
  constexpr int MT = CIN / 16;               // channel tiles of 16
  constexpr int Cs = FIRST ? CIN : 2 * CIN;  // scalar concat channels
  constexpr int Cv = FIRST ? CIN : 3 * CIN;  // vector concat channels

  const int i    = blockIdx.x;
  const int wave = threadIdx.x >> 5;
  const int lane = threadIdx.x & 31;
  const int p    = lane & 15;          // pair index within j-tile
  const int hib  = (lane >> 4) & 1;    // low/high half of wave

  const float* w1s[2] = {w1a, w1b};
  const float* b1s[2] = {b1a, b1b};
  const float* w2s[2] = {w2a, w2b};
  const float* b2s[2] = {b2a, b2b};

  // ---- load weights as WMMA A operands (w^T: row = out channel, K = in) ----
  v16bf A1[2][MT][3];   // dense1: K = 96 (80 rbf + bias slot + pad)
  v16bf A2[2][MT];      // dense2: K = CIN (<=32)
  float B2r[2][MT][8];  // b2 for this lane's channel range
  #pragma unroll
  for (int rad = 0; rad < 2; ++rad){
    #pragma unroll
    for (int mt = 0; mt < MT; ++mt){
      const int row = mt * 16 + p;
      #pragma unroll
      for (int t = 0; t < 3; ++t){
        #pragma unroll
        for (int m = 0; m < 16; ++m){
          // A k order used here (same as Brbf): K = t*32 + hi*16 + m
          int K = t * 32 + hib * 16 + m;
          float v = 0.0f;
          if (K < NRBF)       v = w1s[rad][K * CIN + row];
          else if (K == NRBF) v = b1s[rad][row];          // bias folded in K=80
          A1[rad][mt][t][m] = f2bf(v);
        }
      }
      #pragma unroll
      for (int m = 0; m < 16; ++m){
        int K = hib * 16 + m;
        float v = (K < CIN) ? w2s[rad][K * CIN + row] : 0.0f;
        A2[rad][mt][m] = f2bf(v);
      }
      #pragma unroll
      for (int g = 0; g < 8; ++g)
        B2r[rad][mt][g] = b2s[rad][mt * 16 + hib * 8 + g];
    }
  }

  const float rix = r[i * 3 + 0], riy = r[i * 3 + 1], riz = r[i * 3 + 2];

  // message accumulators (per-lane: its 8 channels, its pair, summed over j)
  float acc_s00[MT][8]    = {};
  float acc_v01[MT][8][3] = {};
  float acc_dot[8]        = {};
  float acc_v10[8][3]     = {};
  float acc_cx [8][3]     = {};

  for (int jt = wave; jt < N_ATOMS / 16; jt += 8){
    const int j = jt * 16 + p;
    float dx = rix - r[j * 3 + 0];
    float dy = riy - r[j * 3 + 1];
    float dz = riz - r[j * 3 + 2];
    float d  = sqrtf(dx * dx + dy * dy + dz * dz + 1e-12f);
    float invd = 1.0f / d;
    float ux = dx * invd, uy = dy * invd, uz = dz * invd;
    float maskv = (j == i) ? 0.0f : 1.0f;

    // ---- RBF in B-layout: lane = pair, K = t*32 + hi*16 + m ----
    v16bf Brbf[3];
    #pragma unroll
    for (int t = 0; t < 3; ++t){
      #pragma unroll
      for (int m = 0; m < 16; ++m){
        int K = t * 32 + hib * 16 + m;
        float v;
        if (K < NRBF){
          float ck = -1.0f + 0.2f * (float)K;
          float dd = d - ck;
          v = __expf(-12.5f * dd * dd);           // exp(-(d-c)^2 / (2*0.2^2))
        } else {
          v = (K == NRBF) ? 1.0f : 0.0f;          // bias slot
        }
        Brbf[t][m] = f2bf(v);
      }
    }

    // ---- neighbor features (vectorized b128 loads; offsets 16B aligned) ----
    float s8[MT][8];
    #pragma unroll
    for (int mt = 0; mt < MT; ++mt){
      const float4* sp = (const float4*)(s_in + j * CIN + mt * 16 + hib * 8);
      float4 sa = sp[0], sb = sp[1];
      s8[mt][0] = sa.x; s8[mt][1] = sa.y; s8[mt][2] = sa.z; s8[mt][3] = sa.w;
      s8[mt][4] = sb.x; s8[mt][5] = sb.y; s8[mt][6] = sb.z; s8[mt][7] = sb.w;
    }

    float vv[8][3];
    if constexpr(!FIRST){
      const float4* vp = (const float4*)(v_in + (j * CIN + hib * 8) * 3);
      float vflat[24];
      #pragma unroll
      for (int q = 0; q < 6; ++q){
        float4 tq = vp[q];
        vflat[q * 4 + 0] = tq.x; vflat[q * 4 + 1] = tq.y;
        vflat[q * 4 + 2] = tq.z; vflat[q * 4 + 3] = tq.w;
      }
      #pragma unroll
      for (int g = 0; g < 8; ++g)
        #pragma unroll
        for (int x = 0; x < 3; ++x)
          vv[g][x] = vflat[g * 3 + x];
    }

    // ---- two radial MLPs via WMMA ----
    float Rs[2][MT][8];
    #pragma unroll
    for (int rad = 0; rad < 2; ++rad){
      v8f h[MT];
      #pragma unroll
      for (int mt = 0; mt < MT; ++mt){
        v8f c = {};
        #pragma unroll
        for (int t = 0; t < 3; ++t)
          c = __builtin_amdgcn_wmma_f32_16x16x32_bf16(
                false, A1[rad][mt][t], false, Brbf[t], (short)0, c, false, false);
        h[mt] = c;
      }
      // ssp activation, then repack D-layout -> B-layout for dense2
      float hs[MT][8], ho[MT][8];
      #pragma unroll
      for (int mt = 0; mt < MT; ++mt)
        #pragma unroll
        for (int g = 0; g < 8; ++g)
          hs[mt][g] = sspf(h[mt][g]);
      #pragma unroll
      for (int mt = 0; mt < MT; ++mt)
        #pragma unroll
        for (int g = 0; g < 8; ++g)
          ho[mt][g] = __shfl_xor(hs[mt][g], 16, 32);

      v16bf Bh;
      #pragma unroll
      for (int m = 0; m < 16; ++m){
        float v;
        if constexpr (CIN == 32){
          v = hib ? ((m < 8) ? ho[1][m] : hs[1][m - 8])
                  : ((m < 8) ? hs[0][m] : ho[0][m - 8]);
        } else {
          v = hib ? 0.0f
                  : ((m < 8) ? hs[0][m] : ho[0][m - 8]);
        }
        Bh[m] = f2bf(v);
      }

      #pragma unroll
      for (int mt = 0; mt < MT; ++mt){
        v8f c = {};
        c = __builtin_amdgcn_wmma_f32_16x16x32_bf16(
              false, A2[rad][mt], false, Bh, (short)0, c, false, false);
        #pragma unroll
        for (int g = 0; g < 8; ++g)
          Rs[rad][mt][g] = (c[g] + B2r[rad][mt][g]) * maskv;
      }
    }

    // ---- tensor-product message accumulation ----
    #pragma unroll
    for (int mt = 0; mt < MT; ++mt)
      #pragma unroll
      for (int g = 0; g < 8; ++g){
        float R0v = Rs[0][mt][g];
        float R1v = Rs[1][mt][g];
        float sj  = s8[mt][g];
        acc_s00[mt][g]    += R0v * sj;                 // 0 x 0 -> 0
        acc_v01[mt][g][0] += R1v * ux * sj;            // 0 x 1 -> 1
        acc_v01[mt][g][1] += R1v * uy * sj;
        acc_v01[mt][g][2] += R1v * uz * sj;
        if constexpr(!FIRST){
          float vx = vv[g][0], vy = vv[g][1], vz = vv[g][2];
          acc_v10[g][0] += R0v * vx;                   // 1 x 0 -> 1
          acc_v10[g][1] += R0v * vy;
          acc_v10[g][2] += R0v * vz;
          acc_dot[g]    += R1v * (ux * vx + uy * vy + uz * vz);   // 1x1 -> 0
          acc_cx[g][0]  += R1v * (uy * vz - uz * vy);             // 1x1 -> 1
          acc_cx[g][1]  += R1v * (uz * vx - ux * vz);
          acc_cx[g][2]  += R1v * (ux * vy - uy * vx);
        }
      }
  }

  // ---- reduce over the 16 pair-lanes in each half, atomicAdd to output ----
  auto redu = [&](float x){
    #pragma unroll
    for (int off = 1; off < 16; off <<= 1) x += __shfl_xor(x, off, 32);
    return x;
  };

  #pragma unroll
  for (int mt = 0; mt < MT; ++mt)
    #pragma unroll
    for (int g = 0; g < 8; ++g){
      int u = mt * 16 + hib * 8 + g;
      float s0r = redu(acc_s00[mt][g]);
      float v0  = redu(acc_v01[mt][g][0]);
      float v1  = redu(acc_v01[mt][g][1]);
      float v2  = redu(acc_v01[mt][g][2]);
      if (p == 0){
        atomicAdd(&out_s[i * Cs + u], s0r);
        atomicAdd(&out_v[(i * Cv + u) * 3 + 0], v0);
        atomicAdd(&out_v[(i * Cv + u) * 3 + 1], v1);
        atomicAdd(&out_v[(i * Cv + u) * 3 + 2], v2);
      }
    }
  if constexpr(!FIRST){
    #pragma unroll
    for (int g = 0; g < 8; ++g){
      int u = hib * 8 + g;
      float dt = redu(acc_dot[g]);
      float a0 = redu(acc_v10[g][0]), a1 = redu(acc_v10[g][1]), a2 = redu(acc_v10[g][2]);
      float c0 = redu(acc_cx[g][0]),  c1 = redu(acc_cx[g][1]),  c2 = redu(acc_cx[g][2]);
      if (p == 0){
        atomicAdd(&out_s[i * Cs + CIN + u], dt);
        atomicAdd(&out_v[(i * Cv + CIN + u) * 3 + 0], a0);
        atomicAdd(&out_v[(i * Cv + CIN + u) * 3 + 1], a1);
        atomicAdd(&out_v[(i * Cv + CIN + u) * 3 + 2], a2);
        atomicAdd(&out_v[(i * Cv + 2 * CIN + u) * 3 + 0], c0);
        atomicAdd(&out_v[(i * Cv + 2 * CIN + u) * 3 + 1], c1);
        atomicAdd(&out_v[(i * Cv + 2 * CIN + u) * 3 + 2], c2);
      }
    }
  }
}

// -------------------------------------------------------------------------
// Self-interaction + equivariant activation (per-atom; tiny).
// -------------------------------------------------------------------------
template<int Cs, int Cv>
__global__ void si_kernel(const float* __restrict__ in_s, const float* __restrict__ in_v,
                          const float* __restrict__ w_s,  const float* __restrict__ b_s,
                          const float* __restrict__ w_v,  const float* __restrict__ a_s,
                          const float* __restrict__ a_v,
                          float* __restrict__ out_s, float* __restrict__ out_v){
  int t = blockIdx.x * blockDim.x + threadIdx.x;
  if (t >= N_ATOMS * 16) return;
  int i = t >> 4, u = t & 15;
  float ss = 0.0f;
  #pragma unroll
  for (int c = 0; c < Cs; ++c) ss += w_s[u * Cs + c] * in_s[i * Cs + c];
  out_s[t] = sspf(ss + b_s[u] + a_s[u]);
  float vx = 0.0f, vy = 0.0f, vz = 0.0f;
  #pragma unroll
  for (int c = 0; c < Cv; ++c){
    float w = w_v[u * Cv + c];
    vx += w * in_v[(i * Cv + c) * 3 + 0];
    vy += w * in_v[(i * Cv + c) * 3 + 1];
    vz += w * in_v[(i * Cv + c) * 3 + 2];
  }
  float nrm  = sqrtf(vx * vx + vy * vy + vz * vz + 1e-12f);
  float gate = sspf(nrm + a_v[u]);
  out_v[t * 3 + 0] = vx * gate;
  out_v[t * 3 + 1] = vy * gate;
  out_v[t * 3 + 2] = vz * gate;
}

// -------------------------------------------------------------------------
// Final energy: sum of all scalar features after conv4.
// -------------------------------------------------------------------------
__global__ void reduce_kernel(const float* __restrict__ s, float* __restrict__ out){
  __shared__ float red[256];
  float acc = 0.0f;
  for (int k = threadIdx.x; k < N_ATOMS * 16; k += 256) acc += s[k];
  red[threadIdx.x] = acc;
  __syncthreads();
  for (int off = 128; off > 0; off >>= 1){
    if (threadIdx.x < off) red[threadIdx.x] += red[threadIdx.x + off];
    __syncthreads();
  }
  if (threadIdx.x == 0) out[0] = red[0];
}

// -------------------------------------------------------------------------
extern "C" void kernel_launch(void* const* d_in, const int* in_sizes, int n_in,
                              void* d_out, int out_size, void* d_ws, size_t ws_size,
                              hipStream_t stream){
  (void)in_sizes; (void)n_in; (void)out_size; (void)ws_size;

  const float* r   = (const float*)d_in[0];
  const int*   z   = (const int*)  d_in[1];
  const float* emb = (const float*)d_in[2];
  // per-conv param order: r0{w1,b1,w2,b2}, r1{w1,b1,w2,b2}, w_s, b_s, w_v, a_s, a_v
  const float* cp[4][13];
  for (int l = 0; l < 4; ++l)
    for (int k = 0; k < 13; ++k)
      cp[l][k] = (const float*)d_in[3 + l * 13 + k];

  float* ws    = (float*)d_ws;
  float* s_cur = ws;                              // [1024*32] feature scalars
  float* v_cur = ws + 1024 * 32;                  // [1024*16*3] feature vectors
  float* o_s   = v_cur + 1024 * 16 * 3;           // [1024*32] raw scalar msgs
  float* o_v   = o_s + 1024 * 32;                 // [1024*48*3] raw vector msgs

  embed_kernel<<<(N_ATOMS * 32 + 255) / 256, 256, 0, stream>>>(emb, z, s_cur);

  // ---- conv1 (CIN=32, first layer: scalars only in) ----
  hipMemsetAsync(o_s, 0, 1024 * 32 * sizeof(float), stream);
  hipMemsetAsync(o_v, 0, 1024 * 32 * 3 * sizeof(float), stream);
  conv_kernel<32, true><<<N_ATOMS, 256, 0, stream>>>(
      r, s_cur, nullptr,
      cp[0][0], cp[0][1], cp[0][2], cp[0][3],
      cp[0][4], cp[0][5], cp[0][6], cp[0][7], o_s, o_v);
  si_kernel<32, 32><<<(N_ATOMS * 16 + 255) / 256, 256, 0, stream>>>(
      o_s, o_v, cp[0][8], cp[0][9], cp[0][10], cp[0][11], cp[0][12], s_cur, v_cur);

  // ---- conv2..conv4 (CIN=16, scalars + vectors in) ----
  for (int l = 1; l < 4; ++l){
    hipMemsetAsync(o_s, 0, 1024 * 32 * sizeof(float), stream);
    hipMemsetAsync(o_v, 0, 1024 * 48 * 3 * sizeof(float), stream);
    conv_kernel<16, false><<<N_ATOMS, 256, 0, stream>>>(
        r, s_cur, v_cur,
        cp[l][0], cp[l][1], cp[l][2], cp[l][3],
        cp[l][4], cp[l][5], cp[l][6], cp[l][7], o_s, o_v);
    si_kernel<32, 48><<<(N_ATOMS * 16 + 255) / 256, 256, 0, stream>>>(
        o_s, o_v, cp[l][8], cp[l][9], cp[l][10], cp[l][11], cp[l][12], s_cur, v_cur);
  }

  reduce_kernel<<<1, 256, 0, stream>>>(s_cur, (float*)d_out);
}
